// MultiHeadAttention_38594576122149
// MI455X (gfx1250) — compile-verified, matrix operand order checked by reading, not run on hardware
//
#include <hip/hip_runtime.h>

#define EMBED 1024
#define HEADS 16
#define HDIM  64
#define SEQ   2048
#define BATCH 4

typedef __attribute__((ext_vector_type(16))) __bf16 v16bf;
typedef __attribute__((ext_vector_type(8)))  float  v8f;
typedef __attribute__((ext_vector_type(4)))  unsigned int v4u;
typedef __attribute__((ext_vector_type(4)))  float  v4f;

// One WMMA A/B operand: 16 bf16 per lane = two contiguous 16-byte runs.
union BF16Frag {
  v16bf v;
  unsigned short u[16];
  v4u q2[2];
};

// ---------------- gfx1250 async copy-to-LDS (guarded) ------------------------
#if __has_builtin(__builtin_amdgcn_global_load_async_to_lds_b128)
#define HAS_ASYNC_LDS 1
// Builtin signature (from hipcc diagnostic): param1 is AS1 pointer to
// gcc-vector 'int __attribute__((vector_size(16)))'; 4 args total.
typedef int v4i_vs __attribute__((vector_size(16)));
typedef __attribute__((address_space(1))) v4i_vs* gptr_v4i;
typedef __attribute__((address_space(3))) v4i_vs* lptr_v4i;
__device__ __forceinline__ void async_cp16(const unsigned short* g, unsigned short* l) {
  __builtin_amdgcn_global_load_async_to_lds_b128(
      (gptr_v4i)(unsigned short*)g, (lptr_v4i)l, 0, 0);
}
#else
#define HAS_ASYNC_LDS 0
__device__ __forceinline__ void async_cp16(const unsigned short* g, unsigned short* l) {
  *(v4u*)l = *(const v4u*)g;   // sync fallback: global_load_b128 + ds_store_b128
}
#endif

template <int N>
__device__ __forceinline__ void wait_asynccnt() {
#if HAS_ASYNC_LDS
#if __has_builtin(__builtin_amdgcn_s_wait_asynccnt)
  __builtin_amdgcn_s_wait_asynccnt((short)N);
#else
  asm volatile("s_wait_asynccnt %0" ::"n"(N) : "memory");
#endif
#endif
}

// float -> bf16 round-to-nearest-even via integer math
__device__ __forceinline__ unsigned short f2bf(float f) {
  unsigned int u = __float_as_uint(f);
  u += 0x7FFFu + ((u >> 16) & 1u);
  return (unsigned short)(u >> 16);
}

__device__ __forceinline__ unsigned int pack2bf(float lo, float hi) {
  return (unsigned int)f2bf(lo) | ((unsigned int)f2bf(hi) << 16);
}

// Per CDNA5 16-bit A/B striping: lane's elements are K = 8*laneHalf + {0..7}
// and K = 16 + 8*laneHalf + {0..7}. p must already include the 8*laneHalf term.
__device__ __forceinline__ void load_frag(BF16Frag& f, const unsigned short* p) {
  f.q2[0] = *(const v4u*)p;
  f.q2[1] = *(const v4u*)(p + 16);
}

// ---------------- bulk f32 -> bf16 conversion (bandwidth-trivial) ------------
__global__ void __launch_bounds__(256)
cvt_f32_bf16(const float* __restrict__ src, unsigned short* __restrict__ dst, long n) {
  const long i = ((long)blockIdx.x * 256 + threadIdx.x) * 8;
  if (i >= n) return;
  const v4f f0 = *(const v4f*)(src + i);
  const v4f f1 = *(const v4f*)(src + i + 4);
  v4u o;
  o.x = pack2bf(f0.x, f0.y);
  o.y = pack2bf(f0.z, f0.w);
  o.z = pack2bf(f1.x, f1.y);
  o.w = pack2bf(f1.z, f1.w);
  *(v4u*)(dst + i) = o;
}

// ---------------- Y[M,N] = A[M,K] @ B[N,K]^T, all-bf16, K = 1024 -------------
// 4 waves/block share one 64-row A stripe (async-staged, double-buffered LDS);
// each wave owns one 64-col B stripe and a 64x64 f32 accumulator (16 WMMA tiles).
// STORE_MODE: 0 = f32 row-major, 1 = bf16 row-major, 2 = bf16 V-transposed [b,h,d,s]
template <int STORE_MODE>
__global__ void __launch_bounds__(128)
gemm_bf16_wmma(const unsigned short* __restrict__ A,
               const unsigned short* __restrict__ B,
               void* __restrict__ Yptr, int M, int N, float outScale) {
  constexpr int K = 1024;
  __shared__ unsigned short As[2][64 * 32];   // 8 KB double buffer
  const int tid  = threadIdx.x;
  const int lane = tid & 31;
  const int wib  = tid >> 5;
  const int tilesN = N >> 6;                  // multiple of 4
  const int wave0  = blockIdx.x * 4;
  const int tm = (wave0 / tilesN) << 6;       // shared by the 4 waves
  const int tn = ((wave0 + wib) % tilesN) << 6;
  if (tm >= M) return;
  const int lh = lane >> 4;
  const int lm = lane & 15;

  // staging assignment: thread covers 32B = two b128 of row (tid>>1)
  const int sRow  = tid >> 1;
  const int sCol  = (tid & 1) * 16;           // element offset within 32-elem row
  const unsigned short* sSrc = A + (long)(tm + sRow) * K + sCol;
  unsigned short* sDst0 = &As[0][sRow * 32 + sCol];
  unsigned short* sDst1 = &As[1][sRow * 32 + sCol];

  const unsigned short* bRow[4];
#pragma unroll
  for (int j = 0; j < 4; ++j) bRow[j] = B + (long)(tn + j * 16 + lm) * K + 8 * lh;

  v8f acc[4][4] = {};

  // prologue: stage tile 0 into buffer 0
  async_cp16(sSrc, sDst0);
  async_cp16(sSrc + 8, sDst0 + 8);

#pragma unroll 1
  for (int k0 = 0; k0 < K; k0 += 32) {
    const unsigned short* aBuf = As[(k0 >> 5) & 1];
    if (k0 + 32 < K) {
      unsigned short* nxt = (((k0 >> 5) + 1) & 1) ? sDst1 : sDst0;
      const unsigned short* gs = sSrc + k0 + 32;
      async_cp16(gs, nxt);
      async_cp16(gs + 8, nxt + 8);
      wait_asynccnt<2>();                     // current tile's 2 ops complete
    } else {
      wait_asynccnt<0>();
    }
    __syncthreads();                          // all waves' stripes visible

    BF16Frag a[4];
#pragma unroll
    for (int i = 0; i < 4; ++i) {             // ds_load_b128 x2 per fragment
      const unsigned short* ap = aBuf + (i * 16 + lm) * 32 + 8 * lh;
      a[i].q2[0] = *(const v4u*)ap;
      a[i].q2[1] = *(const v4u*)(ap + 16);
    }
#pragma unroll
    for (int j = 0; j < 4; ++j) {
      BF16Frag b;
      load_frag(b, bRow[j] + k0);             // global_load_b128 x2
#pragma unroll
      for (int i = 0; i < 4; ++i)
        acc[i][j] = __builtin_amdgcn_wmma_f32_16x16x32_bf16(
            false, a[i].v, false, b.v, (short)0, acc[i][j], false, false);
    }
    __syncthreads();                          // done reading before next overwrite
  }

#pragma unroll
  for (int i = 0; i < 4; ++i)
#pragma unroll
    for (int j = 0; j < 4; ++j)
#pragma unroll
      for (int r = 0; r < 8; ++r) {
        const int row = tm + i * 16 + r + 8 * lh;   // C/D layout: VGPR r, lane-half
        const int col = tn + j * 16 + lm;
        if (STORE_MODE == 0) {
          ((float*)Yptr)[(long)row * N + col] = acc[i][j][r];
        } else if (STORE_MODE == 1) {
          ((unsigned short*)Yptr)[(long)row * N + col] = f2bf(acc[i][j][r] * outScale);
        } else {
          const int bb = row >> 11, t = row & (SEQ - 1);
          const int hh = col >> 6,  d = col & (HDIM - 1);
          ((unsigned short*)Yptr)[((long)(bb * HEADS + hh) * HDIM + d) * SEQ + t] =
              f2bf(acc[i][j][r] * outScale);
        }
      }
}

// ---------------- flash attention: one wave = 16 query rows of one (b,h) ----
// Q pre-scaled by 1/sqrt(64). Q,K bf16 [b,s,h*64+d]; V bf16 transposed [b,h,d,s].
__global__ void __launch_bounds__(128)
attn_wmma(const unsigned short* __restrict__ Q,
          const unsigned short* __restrict__ Km,
          const unsigned short* __restrict__ Vt,
          unsigned short* __restrict__ O) {
  __shared__ unsigned short ldsP[4 * 16 * 32];
  const int lane = threadIdx.x & 31;
  const int wib  = threadIdx.x >> 5;
  const int wave = blockIdx.x * 4 + wib;
  const int qTiles = SEQ / 16;                 // 128
  const int qt = wave % qTiles;
  const int h  = (wave / qTiles) % HEADS;
  const int b  = wave / (qTiles * HEADS);
  const int lh = lane >> 4, lm = lane & 15;
  unsigned short* pl = ldsP + wib * 16 * 32;   // per-wave tile; same-wave DS is in-order
  const long headOff = (long)h * HDIM;

  BF16Frag qf[2];
#pragma unroll
  for (int kk = 0; kk < 2; ++kk)
    load_frag(qf[kk], Q + (long)(b * SEQ + qt * 16 + lm) * EMBED + headOff + kk * 32 + 8 * lh);

  v8f oacc[4] = {};
  float mrow[8], lrow[8];
#pragma unroll
  for (int r = 0; r < 8; ++r) { mrow[r] = -3.0e38f; lrow[r] = 0.0f; }

  const unsigned short* vBase = Vt + (long)(b * HEADS + h) * HDIM * SEQ;

  for (int ks = 0; ks < SEQ; ks += 32) {
    // ---- scores S[16x32] = Q @ K^T (Q pre-scaled; contraction over d) ----
    v8f sc[2] = {};
#pragma unroll
    for (int t = 0; t < 2; ++t)
#pragma unroll
      for (int kk = 0; kk < 2; ++kk) {
        BF16Frag kf;  // B-frag: lane = key column, elements walk d (contiguous)
        load_frag(kf, Km + (long)(b * SEQ + ks + t * 16 + lm) * EMBED + headOff + kk * 32 + 8 * lh);
        sc[t] = __builtin_amdgcn_wmma_f32_16x16x32_bf16(
            false, qf[kk].v, false, kf.v, (short)0, sc[t], false, false);
      }

    // ---- online softmax (rows live across the 16 lanes of a lane-half) ----
    float alpha[8];
#pragma unroll
    for (int r = 0; r < 8; ++r) {
      float rm = fmaxf(sc[0][r], sc[1][r]);
#pragma unroll
      for (int off = 1; off < 16; off <<= 1)
        rm = fmaxf(rm, __shfl_xor(rm, off, 32));
      const float mn = fmaxf(mrow[r], rm);
      alpha[r] = __expf(mrow[r] - mn);
      const float p0 = __expf(sc[0][r] - mn);
      const float p1 = __expf(sc[1][r] - mn);
      sc[0][r] = p0; sc[1][r] = p1;
      float rs = p0 + p1;
#pragma unroll
      for (int off = 1; off < 16; off <<= 1)
        rs += __shfl_xor(rs, off, 32);
      lrow[r] = lrow[r] * alpha[r] + rs;
      mrow[r] = mn;
    }
#pragma unroll
    for (int j = 0; j < 4; ++j)
#pragma unroll
      for (int r = 0; r < 8; ++r) oacc[j][r] *= alpha[r];

    // ---- relayout P: C-layout -> LDS -> A-fragment (ds_load_b128 reads) ----
#pragma unroll
    for (int t = 0; t < 2; ++t)
#pragma unroll
      for (int r = 0; r < 8; ++r)
        pl[(r + 8 * lh) * 32 + t * 16 + lm] = f2bf(sc[t][r]);
    BF16Frag pf;
    {
      const unsigned short* pp = pl + lm * 32 + 8 * lh;
      pf.q2[0] = *(const v4u*)pp;
      pf.q2[1] = *(const v4u*)(pp + 16);
    }

    // ---- O += P @ V (contraction over 32 keys; Vt makes keys contiguous) ----
#pragma unroll
    for (int j = 0; j < 4; ++j) {
      BF16Frag vf;  // B-frag: lane = d column, elements walk key (contiguous in Vt)
      load_frag(vf, vBase + (long)(j * 16 + lm) * SEQ + ks + 8 * lh);
      oacc[j] = __builtin_amdgcn_wmma_f32_16x16x32_bf16(
          false, pf.v, false, vf.v, (short)0, oacc[j], false, false);
    }
  }

  // ---- finalize: divide by softmax denominator, store bf16 ----
#pragma unroll
  for (int j = 0; j < 4; ++j)
#pragma unroll
    for (int r = 0; r < 8; ++r) {
      const int row = qt * 16 + r + 8 * lh;
      O[(long)(b * SEQ + row) * EMBED + headOff + j * 16 + lm] =
          f2bf(oacc[j][r] / lrow[r]);
    }
}

extern "C" void kernel_launch(void* const* d_in, const int* in_sizes, int n_in,
                              void* d_out, int out_size, void* d_ws, size_t ws_size,
                              hipStream_t stream) {
  (void)in_sizes; (void)n_in; (void)out_size; (void)ws_size;
  const float* x  = (const float*)d_in[0];
  const float* wq = (const float*)d_in[1];
  const float* wk = (const float*)d_in[2];
  const float* wv = (const float*)d_in[3];
  const float* wo = (const float*)d_in[4];

  const int  M     = BATCH * SEQ;           // 8192
  const long plane = (long)M * EMBED;       // 8.4M elems (16 MB bf16)
  const long wsz   = (long)EMBED * EMBED;   // 1M elems (2 MB bf16)

  unsigned short* xb  = (unsigned short*)d_ws;
  unsigned short* wqb = xb  + plane;
  unsigned short* wkb = wqb + wsz;
  unsigned short* wvb = wkb + wsz;
  unsigned short* wob = wvb + wsz;
  unsigned short* q   = wob + wsz;
  unsigned short* k   = q   + plane;
  unsigned short* vt  = k   + plane;
  unsigned short* a   = vt  + plane;        // total ~88 MB

  // f32 -> bf16 staging (one pass; everything downstream uses b128 loads)
  cvt_f32_bf16<<<(int)(plane / 8 / 256), 256, 0, stream>>>(x,  xb,  plane);
  cvt_f32_bf16<<<(int)(wsz   / 8 / 256), 256, 0, stream>>>(wq, wqb, wsz);
  cvt_f32_bf16<<<(int)(wsz   / 8 / 256), 256, 0, stream>>>(wk, wkb, wsz);
  cvt_f32_bf16<<<(int)(wsz   / 8 / 256), 256, 0, stream>>>(wv, wvb, wsz);
  cvt_f32_bf16<<<(int)(wsz   / 8 / 256), 256, 0, stream>>>(wo, wob, wsz);

  const int gemmBlocks = (M / 64) * (EMBED / 64) / 4;     // 512 blocks x 4 waves
  gemm_bf16_wmma<1><<<gemmBlocks, 128, 0, stream>>>(xb, wqb, q,  M, EMBED, 0.125f);
  gemm_bf16_wmma<1><<<gemmBlocks, 128, 0, stream>>>(xb, wkb, k,  M, EMBED, 1.0f);
  gemm_bf16_wmma<2><<<gemmBlocks, 128, 0, stream>>>(xb, wvb, vt, M, EMBED, 1.0f);

  const int attnBlocks = BATCH * HEADS * (SEQ / 16) / 4;  // 2048 blocks x 4 waves
  attn_wmma<<<attnBlocks, 128, 0, stream>>>(q, k, vt, a);

  gemm_bf16_wmma<0><<<gemmBlocks, 128, 0, stream>>>(a, wob, d_out, M, EMBED, 1.0f);
}
